// CodeBook_86457691669012
// MI455X (gfx1250) — compile-verified, matrix operand order checked by reading, not run on hardware
//
#include <hip/hip_runtime.h>
#include <hip/hip_bf16.h>

// Problem sizes (B=8, S=4096, D=512, K=4096)
#define NROWS   32768     // B*S
#define DIM     512
#define KCODES  4096
#define M_TILE  64        // x rows per workgroup
#define C_TILE  32        // codes per inner tile
#define THREADS 256       // 8 waves (wave32): 4 m-subtiles x 2 n-subtiles
#define NTILES  (KCODES / C_TILE)
#define TILE_US (C_TILE * DIM)            // ushorts per (hi or lo) code tile
#define TILE_Q  (TILE_US / 8)             // uint4 (16B) chunks per tile = 2048
#define Q_PER_T (TILE_Q / THREADS)        // 8 b128 transfers per thread per tile

typedef __attribute__((ext_vector_type(16))) __bf16 v16bf;
typedef __attribute__((ext_vector_type(8)))  float  v8f;
typedef int v4i __attribute__((vector_size(16)));   // matches builtin param type

// CDNA5 async global->LDS path (ASYNCcnt), guarded so the build can't regress.
#if defined(__gfx1250__) && __has_builtin(__builtin_amdgcn_global_load_async_to_lds_b128)
#define USE_ASYNC_LDS 1
// C-style casts: const-strip + addrspacecast (generic -> AS1 / AS3).
#define GAS(p) ((__attribute__((address_space(1))) v4i*)(p))
#define LAS(p) ((__attribute__((address_space(3))) v4i*)(p))
#if __has_builtin(__builtin_amdgcn_s_wait_asynccnt)
#define WAIT_ASYNC(n) __builtin_amdgcn_s_wait_asynccnt(n)
#else
#define WAIT_ASYNC(n)                                              \
  do {                                                             \
    if ((n) == 0) asm volatile("s_wait_asynccnt 0x0" ::: "memory");\
    else asm volatile("s_wait_asynccnt 0x10" ::: "memory");        \
  } while (0)
#endif
#endif

__device__ __forceinline__ unsigned short bf16_rne(float f) {
  unsigned int u = __float_as_uint(f);
  u += 0x7FFFu + ((u >> 16) & 1u);
  return (unsigned short)(u >> 16);
}
__device__ __forceinline__ float bf16_to_f32(unsigned short h) {
  return __uint_as_float(((unsigned int)h) << 16);
}

// A/B fragment fetch from LDS for v_wmma_*_16x16x32_bf16.
// 16-bit 16x32 layout: lanes 0-15 hold K=0..7 (VGPR0-3) and K=16..23 (VGPR4-7);
// lanes 16-31 hold K=8..15 and K=24..31. Row/col = lane&15.
__device__ __forceinline__ v16bf frag_lds(const unsigned short* rowp, int kb, int half) {
  union { v16bf v; uint4 q[2]; } f;
  f.q[0] = *(const uint4*)(rowp + kb + 8 * half);
  f.q[1] = *(const uint4*)(rowp + kb + 16 + 8 * half);
  return f.v;
}

__device__ __forceinline__ v8f wmma_bf16(v16bf a, v16bf b, v8f c) {
  return __builtin_amdgcn_wmma_f32_16x16x32_bf16(false, a, false, b, (short)0, c,
                                                 false, false);
}

// ---------------------------------------------------------------------------
// Kernel 1: codebook -> bf16 hi/lo split + exact f32 ||c||^2; zero loss slots.
// ---------------------------------------------------------------------------
__global__ void __launch_bounds__(128) cb_prep(
    const float* __restrict__ cb,
    unsigned short* __restrict__ cbh,
    unsigned short* __restrict__ cbl,
    float* __restrict__ csq,
    float* __restrict__ out_loss) {
  const int code = blockIdx.x;
  const int tid  = threadIdx.x;
  const float* row = cb + (size_t)code * DIM;
  float s = 0.f;
#pragma unroll
  for (int j = 0; j < DIM / 128; ++j) {
    int d = tid + j * 128;
    float f = row[d];
    unsigned short h = bf16_rne(f);
    cbh[(size_t)code * DIM + d] = h;
    cbl[(size_t)code * DIM + d] = bf16_rne(f - bf16_to_f32(h));
    s += f * f;
  }
#pragma unroll
  for (int o = 16; o >= 1; o >>= 1) s += __shfl_xor(s, o, 32);
  __shared__ float ws[4];
  if ((tid & 31) == 0) ws[tid >> 5] = s;
  __syncthreads();
  if (tid == 0) {
    float t = 0.f;
#pragma unroll
    for (int w = 0; w < 4; ++w) t += ws[w];
    csq[code] = t;
    if (code == 0) { out_loss[0] = 0.f; out_loss[1] = 0.f; }
  }
}

// ---------------------------------------------------------------------------
// Kernel 2: fused distance-GEMM (split-bf16 WMMA, double-buffered async
// global->LDS code-tile staging) + argmin + gather + loss.
// ---------------------------------------------------------------------------
__global__ void __launch_bounds__(THREADS, 1) vq_main(
    const float* __restrict__ x,
    const float* __restrict__ cb,
    const unsigned short* __restrict__ cbh,
    const unsigned short* __restrict__ cbl,
    const float* __restrict__ csq,
    float* __restrict__ out_enc,
    float* __restrict__ out_loss,
    float* __restrict__ out_near) {
  extern __shared__ char smem_raw[];
  unsigned short* xh = (unsigned short*)smem_raw;          // M_TILE*DIM
  unsigned short* xl = xh + M_TILE * DIM;                  // M_TILE*DIM
  unsigned short* ch = xl + M_TILE * DIM;                  // 2 * TILE_US (dbuf)
  unsigned short* cl = ch + 2 * TILE_US;                   // 2 * TILE_US (dbuf)
  float* xsq  = (float*)(cl + 2 * TILE_US);                // M_TILE
  float* redv = xsq + M_TILE;                              // 2*M_TILE
  int*   redi = (int*)(redv + 2 * M_TILE);                 // 2*M_TILE
  int*   fidx = redi + 2 * M_TILE;                         // M_TILE
  float* wls  = (float*)(fidx + M_TILE);                   // 8

  const int tid  = threadIdx.x;
  const int lane = tid & 31;
  const int wave = tid >> 5;      // 0..7
  const int wm   = wave >> 1;     // m-subtile 0..3
  const int wn   = wave & 1;      // n-subtile 0..1
  const int half = lane >> 4;
  const int mc   = lane & 15;
  const int rowBase = blockIdx.x * M_TILE;

  // ---- Phase A: x tile -> bf16 hi/lo in LDS + per-row ||x||^2 -------------
  if (tid < M_TILE) xsq[tid] = 0.f;
  __syncthreads();
  {
    const int r   = tid >> 2;          // 0..63
    const int seg = tid & 3;           // 128-col segment
    const float4* xr4 = (const float4*)(x + (size_t)(rowBase + r) * DIM + seg * 128);
    unsigned short* xhr = xh + r * DIM + seg * 128;
    unsigned short* xlr = xl + r * DIM + seg * 128;
    float s = 0.f;
#pragma unroll 4
    for (int j = 0; j < 32; ++j) {
      float4 f4 = xr4[j];
      const float fv[4] = {f4.x, f4.y, f4.z, f4.w};
#pragma unroll
      for (int q = 0; q < 4; ++q) {
        float f = fv[q];
        unsigned short h = bf16_rne(f);
        xhr[j * 4 + q] = h;
        xlr[j * 4 + q] = bf16_rne(f - bf16_to_f32(h));
        s += f * f;
      }
    }
    atomicAdd(&xsq[r], s);   // ds_add_f32
  }

  // ---- code-tile staging helpers ------------------------------------------
  // Each thread moves Q_PER_T b128 chunks of the hi tile and of the lo tile.
#ifdef USE_ASYNC_LDS
  auto issue_tile = [&](int ct) {
    const int buf = ct & 1;
    const uint4* gh = (const uint4*)(cbh + (size_t)ct * TILE_US);
    const uint4* gl = (const uint4*)(cbl + (size_t)ct * TILE_US);
    uint4* sh = (uint4*)(ch + buf * TILE_US);
    uint4* sl = (uint4*)(cl + buf * TILE_US);
#pragma unroll
    for (int j = 0; j < Q_PER_T; ++j) {
      const int k = tid + j * THREADS;
      __builtin_amdgcn_global_load_async_to_lds_b128(GAS(gh + k), LAS(sh + k), 0, 0);
      __builtin_amdgcn_global_load_async_to_lds_b128(GAS(gl + k), LAS(sl + k), 0, 0);
    }
  };
#else
  auto issue_tile = [&](int ct) {
    const int buf = ct & 1;
    const uint4* gh = (const uint4*)(cbh + (size_t)ct * TILE_US);
    const uint4* gl = (const uint4*)(cbl + (size_t)ct * TILE_US);
    uint4* sh = (uint4*)(ch + buf * TILE_US);
    uint4* sl = (uint4*)(cl + buf * TILE_US);
#pragma unroll
    for (int j = 0; j < Q_PER_T; ++j) {
      const int k = tid + j * THREADS;
      sh[k] = gh[k];
      sl[k] = gl[k];
    }
  };
#endif

  // ---- Phase B: double-buffered loop over code tiles ----------------------
  const unsigned short* a_h = xh + (size_t)(wm * 16 + mc) * DIM;
  const unsigned short* a_l = xl + (size_t)(wm * 16 + mc) * DIM;

  float minv[8];
  int   mini[8];
#pragma unroll
  for (int r = 0; r < 8; ++r) { minv[r] = 3.0e38f; mini[r] = 0; }

  issue_tile(0);                 // prologue fill of buffer 0

  for (int ct = 0; ct < NTILES; ++ct) {
    // Start next tile into the other buffer (safe: it was last read in
    // iteration ct-1, which ended with a barrier).
    if (ct + 1 < NTILES) {
      issue_tile(ct + 1);
#ifdef USE_ASYNC_LDS
      WAIT_ASYNC(2 * Q_PER_T);   // tile ct done (async loads complete in order)
#endif
    } else {
#ifdef USE_ASYNC_LDS
      WAIT_ASYNC(0);
#endif
    }
    __syncthreads();             // tile ct visible to all waves

    const unsigned short* b_h = ch + (ct & 1) * TILE_US + (size_t)(wn * 16 + mc) * DIM;
    const unsigned short* b_l = cl + (ct & 1) * TILE_US + (size_t)(wn * 16 + mc) * DIM;

    // dot(x_i, c_j) ~= xh*ch + xh*cl + xl*ch  (error-compensated bf16)
    v8f acc = {};
#pragma unroll
    for (int kb = 0; kb < DIM; kb += 32) {
      v16bf ah = frag_lds(a_h, kb, half);
      v16bf bh = frag_lds(b_h, kb, half);
      v16bf al = frag_lds(a_l, kb, half);
      v16bf bl = frag_lds(b_l, kb, half);
      acc = wmma_bf16(ah, bh, acc);
      acc = wmma_bf16(ah, bl, acc);
      acc = wmma_bf16(al, bh, acc);
    }

    const int code = ct * C_TILE + wn * 16 + mc;
    const float cs = csq[code];
#pragma unroll
    for (int r = 0; r < 8; ++r) {
      const int rl = wm * 16 + r + 8 * half;   // C/D layout: M = r + 8*(lane/16)
      const float dist = xsq[rl] - 2.0f * acc[r] + cs;
      if (dist < minv[r]) { minv[r] = dist; mini[r] = code; }
    }
    __syncthreads();             // all waves done reading buffer (ct&1)
  }

  // ---- Phase C: argmin reduction ------------------------------------------
#pragma unroll
  for (int r = 0; r < 8; ++r) {
    float mv = minv[r];
    int   mi = mini[r];
#pragma unroll
    for (int o = 1; o < 16; o <<= 1) {   // butterfly within the 16-lane half
      float ov = __shfl_xor(mv, o, 32);
      int   oi = __shfl_xor(mi, o, 32);
      if (ov < mv || (ov == mv && oi < mi)) { mv = ov; mi = oi; }
    }
    if ((lane & 15) == 0) {
      const int rl = wm * 16 + r + 8 * half;
      redv[wn * M_TILE + rl] = mv;
      redi[wn * M_TILE + rl] = mi;
    }
  }
  __syncthreads();
  if (tid < M_TILE) {
    const float v0 = redv[tid], v1 = redv[M_TILE + tid];
    const int   i0 = redi[tid], i1 = redi[M_TILE + tid];
    const int best = (v1 < v0 || (v1 == v0 && i1 < i0)) ? i1 : i0;
    fidx[tid] = best;
    out_enc[rowBase + tid] = (float)best;
  }
  __syncthreads();

  // ---- Phase D: gather nearest (f32 codebook) + MSE loss ------------------
  float ls = 0.f;
  {
    const float4* cb4 = (const float4*)cb;
    const float4* x4  = (const float4*)x;
    float4* on4 = (float4*)out_near;
    for (int i = tid; i < M_TILE * (DIM / 4); i += THREADS) {
      const int r  = i >> 7;            // / (DIM/4)
      const int d4 = i & 127;
      const int idx = fidx[r];
      float4 c = cb4[(size_t)idx * (DIM / 4) + d4];
      float4 xv = x4[(size_t)(rowBase + r) * (DIM / 4) + d4];
      float dx = xv.x - c.x, dy = xv.y - c.y, dz = xv.z - c.z, dw = xv.w - c.w;
      ls += dx * dx + dy * dy + dz * dz + dw * dw;
      on4[(size_t)(rowBase + r) * (DIM / 4) + d4] = c;
    }
  }
#pragma unroll
  for (int o = 16; o >= 1; o >>= 1) ls += __shfl_xor(ls, o, 32);
  if (lane == 0) wls[wave] = ls;
  __syncthreads();
  if (tid == 0) {
    float t = 0.f;
#pragma unroll
    for (int w = 0; w < THREADS / 32; ++w) t += wls[w];
    t *= (1.0f / ((float)NROWS * (float)DIM));
    atomicAdd(&out_loss[0], t);   // codebook_loss
    atomicAdd(&out_loss[1], t);   // encoder_loss (identical value)
  }
}

// ---------------------------------------------------------------------------
extern "C" void kernel_launch(void* const* d_in, const int* in_sizes, int n_in,
                              void* d_out, int out_size, void* d_ws, size_t ws_size,
                              hipStream_t stream) {
  const float* x  = (const float*)d_in[0];   // (8, 4096, 512) f32
  const float* cb = (const float*)d_in[1];   // (4096, 512) f32

  float* out      = (float*)d_out;
  float* out_enc  = out;                     // 32768 (indices as float)
  float* out_loss = out + NROWS;             // [codebook_loss, encoder_loss]
  float* out_near = out + NROWS + 2;         // 32768*512

  // Workspace: cb_hi | cb_lo | csq  (~8.02 MB)
  unsigned short* cbh = (unsigned short*)d_ws;
  unsigned short* cbl = cbh + (size_t)KCODES * DIM;
  float*          csq = (float*)(cbl + (size_t)KCODES * DIM);

  cb_prep<<<KCODES, 128, 0, stream>>>(cb, cbh, cbl, csq, out_loss);

  const size_t smem =
      (size_t)(2 * M_TILE * DIM + 4 * TILE_US) * sizeof(unsigned short) +
      (size_t)M_TILE * sizeof(float) +            // xsq
      (size_t)2 * M_TILE * sizeof(float) +        // redv
      (size_t)2 * M_TILE * sizeof(int) +          // redi
      (size_t)M_TILE * sizeof(int) +              // fidx
      (THREADS / 32) * sizeof(float);             // wls
  vq_main<<<NROWS / M_TILE, THREADS, smem, stream>>>(
      x, cb, cbh, cbl, csq, out_enc, out_loss, out_near);
}
